// RNNDecoder_3925600108955
// MI455X (gfx1250) — compile-verified, matrix operand order checked by reading, not run on hardware
//
#include <hip/hip_runtime.h>
#include <hip/hip_bf16.h>
#include <math.h>
#include <stdint.h>

// Model dims (fixed by the reference)
#define DMODEL 512
#define DKDIM  64
#define BATCH  64
#define TSTEPS 64
#define LNODES 500
#define ENCIN  768
#define DHALF  (DMODEL / 2)

typedef _Float16 half_t;
typedef __attribute__((ext_vector_type(16))) _Float16 v16h;
typedef __attribute__((ext_vector_type(8)))  _Float16 v8h;
typedef __attribute__((ext_vector_type(8)))  float    v8f;

#define LDH 40  // padded LDS row stride in halves (80B) -> conflict-free frag loads
#define TILE_HALVES (64 * LDH)

// ---------------------------------------------------------------------------
// WMMA fragment load: 16-bit A/B 16x16x32 layout (ISA 7.12.2):
//   lanes 0-15 : M/N = lane, K = 0..7  (lo) and 16..23 (hi)
//   lanes 16-31: M/N = lane-16, K = 8..15 (lo) and 24..31 (hi)
__device__ inline v16h load_frag(const half_t* p) {
  v8h lo = *(const v8h*)(p);
  v8h hi = *(const v8h*)(p + 16);
  v16h r;
#pragma unroll
  for (int i = 0; i < 8; ++i) { r[i] = lo[i]; r[8 + i] = hi[i]; }
  return r;
}

// Stage one 64x32 tile, converting f32 -> f16 into padded LDS (sync path).
__device__ inline void stage_tile(half_t* dst, const float* __restrict__ src,
                                  int ld, int row0, int col0) {
  int tid = threadIdx.x;
  int r = tid >> 2;            // 0..63
  int c = (tid & 3) * 8;       // 0,8,16,24
  const float* s = src + (long)(row0 + r) * ld + col0 + c;
  half_t* d = dst + r * LDH + c;
#pragma unroll
  for (int i = 0; i < 8; ++i) d[i] = (half_t)s[i];
}

// Async stage of one 64x32 f16 tile: each of 256 lanes pulls 16B straight into
// LDS via the CDNA5 async global->LDS path (no VGPR round trip, ASYNCcnt).
__device__ inline void stage_async(half_t* dst, const half_t* __restrict__ src,
                                   int ld, int row0, int col0) {
  int tid = threadIdx.x;
  int r = tid >> 2;
  int c = (tid & 3) * 8;
  const half_t* g = src + (long)(row0 + r) * ld + col0 + c;
  // Generic-pointer low 32 bits == LDS byte offset (ISA 10.2 aperture rules).
  unsigned lds = (unsigned)(uintptr_t)(dst + r * LDH + c);
  unsigned long long ga = (unsigned long long)(uintptr_t)g;
  asm volatile("global_load_async_to_lds_b128 %0, %1, off"
               :: "v"(lds), "v"(ga) : "memory");
}

__device__ inline void wait_async0() {
  asm volatile("s_wait_asynccnt 0x0" ::: "memory");
}

// ---------------------------------------------------------------------------
// Sync-staging WMMA GEMM (f32 inputs, converted in LDS staging):
// C[M,N] = epi(A[M,K] @ Bw[N,K]^T + bias[N]);  M%64==0, N%64==0, K%32==0.
// EPI: 0 = identity, 1 = tanh
template <int EPI>
__global__ __launch_bounds__(256) void wmma_gemm_kernel(
    const float* __restrict__ A, const float* __restrict__ Bw,
    const float* __restrict__ bias, float* __restrict__ C,
    int M, int N, int K) {
  __shared__ __align__(16) half_t As[TILE_HALVES];
  __shared__ __align__(16) half_t Bs[TILE_HALVES];

  const int nBase = blockIdx.x * 64;
  const int mBase = blockIdx.y * 64;
  const int wid  = threadIdx.x >> 5;
  const int lane = threadIdx.x & 31;
  const int wr = wid >> 2;
  const int wc = wid & 3;

  v8f acc0 = {};
  v8f acc1 = {};
  const int rrow = lane & 15;
  const int koff = (lane >> 4) * 8;

  for (int k0 = 0; k0 < K; k0 += 32) {
    __syncthreads();
    stage_tile(As, A,  K, mBase, k0);
    stage_tile(Bs, Bw, K, nBase, k0);
    __syncthreads();

    const half_t* ap0 = As + (wr * 32 + rrow) * LDH + koff;
    const half_t* bp  = Bs + (wc * 16 + rrow) * LDH + koff;
    v16h a0 = load_frag(ap0);
    v16h a1 = load_frag(ap0 + 16 * LDH);
    v16h bb = load_frag(bp);
    acc0 = __builtin_amdgcn_wmma_f32_16x16x32_f16(false, a0, false, bb,
                                                  (short)0, acc0, false, false);
    acc1 = __builtin_amdgcn_wmma_f32_16x16x32_f16(false, a1, false, bb,
                                                  (short)0, acc1, false, false);
  }

  const int n = nBase + wc * 16 + (lane & 15);
  const int mrow = mBase + wr * 32 + ((lane >> 4) << 3);
  const float bv = bias ? bias[n] : 0.0f;
#pragma unroll
  for (int r = 0; r < 8; ++r) {
    float x0 = acc0[r] + bv;
    float x1 = acc1[r] + bv;
    if (EPI == 1) { x0 = tanhf(x0); x1 = tanhf(x1); }
    C[(long)(mrow + r) * N + n]      = x0;
    C[(long)(mrow + 16 + r) * N + n] = x1;
  }
}

// ---------------------------------------------------------------------------
// Async double-buffered WMMA GEMM (f16 inputs in memory). Optionally also
// writes an f16 mirror of C (Ch) for downstream bandwidth-bound consumers.
__global__ __launch_bounds__(256) void wmma_gemm_h_kernel(
    const half_t* __restrict__ A, const half_t* __restrict__ Bw,
    const float* __restrict__ bias, float* __restrict__ C,
    half_t* __restrict__ Ch, int M, int N, int K) {
  __shared__ __align__(16) half_t As[2 * TILE_HALVES];
  __shared__ __align__(16) half_t Bs[2 * TILE_HALVES];

  const int nBase = blockIdx.x * 64;
  const int mBase = blockIdx.y * 64;
  const int wid  = threadIdx.x >> 5;
  const int lane = threadIdx.x & 31;
  const int wr = wid >> 2;
  const int wc = wid & 3;

  v8f acc0 = {};
  v8f acc1 = {};
  const int rrow = lane & 15;
  const int koff = (lane >> 4) * 8;

  // Prime buffer 0.
  stage_async(As, A,  K, mBase, 0);
  stage_async(Bs, Bw, K, nBase, 0);

  int buf = 0;
  for (int k0 = 0; k0 < K; k0 += 32) {
    wait_async0();        // this wave's slice of buf is in LDS
    __syncthreads();      // everyone's slice is in LDS

    const int nbuf = buf ^ 1;
    if (k0 + 32 < K) {    // prefetch next K-tile while we do math
      stage_async(As + nbuf * TILE_HALVES, A,  K, mBase, k0 + 32);
      stage_async(Bs + nbuf * TILE_HALVES, Bw, K, nBase, k0 + 32);
    }

    const half_t* ap0 = As + buf * TILE_HALVES + (wr * 32 + rrow) * LDH + koff;
    const half_t* bp  = Bs + buf * TILE_HALVES + (wc * 16 + rrow) * LDH + koff;
    v16h a0 = load_frag(ap0);
    v16h a1 = load_frag(ap0 + 16 * LDH);
    v16h bb = load_frag(bp);
    acc0 = __builtin_amdgcn_wmma_f32_16x16x32_f16(false, a0, false, bb,
                                                  (short)0, acc0, false, false);
    acc1 = __builtin_amdgcn_wmma_f32_16x16x32_f16(false, a1, false, bb,
                                                  (short)0, acc1, false, false);

    __syncthreads();      // all reads of buf done before it is restaged
    buf = nbuf;
  }

  const int n = nBase + wc * 16 + (lane & 15);
  const int mrow = mBase + wr * 32 + ((lane >> 4) << 3);
  const float bv = bias ? bias[n] : 0.0f;
#pragma unroll
  for (int r = 0; r < 8; ++r) {
    float x0 = acc0[r] + bv;
    float x1 = acc1[r] + bv;
    C[(long)(mrow + r) * N + n]      = x0;
    C[(long)(mrow + 16 + r) * N + n] = x1;
    if (Ch) {
      Ch[(long)(mrow + r) * N + n]      = (half_t)x0;
      Ch[(long)(mrow + 16 + r) * N + n] = (half_t)x1;
    }
  }
}

// ---------------------------------------------------------------------------
// Elementwise / small kernels
__global__ void zero_kernel(float* p, int n) {
  int i = blockIdx.x * blockDim.x + threadIdx.x;
  if (i < n) p[i] = 0.0f;
}

__global__ void copy_kernel(const float* __restrict__ s, float* __restrict__ d, int n) {
  int i = blockIdx.x * blockDim.x + threadIdx.x;
  if (i < n) d[i] = s[i];
}

__global__ void cast_h_kernel(const float* __restrict__ s, half_t* __restrict__ d, long n) {
  long i = (long)blockIdx.x * blockDim.x + threadIdx.x;
  if (i < n) d[i] = (half_t)s[i];
}

// emb[t,b,:] = word_emb[tgt_seq[b,t]]
__global__ void embed_kernel(const int* __restrict__ tgt, const float* __restrict__ wemb,
                             float* __restrict__ emb, int n) {
  int i = blockIdx.x * blockDim.x + threadIdx.x;
  if (i >= n) return;
  int d = i % DMODEL;
  int r = i / DMODEL;
  int b = r % BATCH;
  int t = r / BATCH;
  int tok = tgt[b * TSTEPS + t];
  emb[i] = wemb[(long)tok * DMODEL + d];
}

__global__ void concat2_kernel(const float* __restrict__ a, const float* __restrict__ b,
                               float* __restrict__ out, int da, int db) {
  int n = BATCH * (da + db);
  int i = blockIdx.x * blockDim.x + threadIdx.x;
  if (i >= n) return;
  int w = da + db;
  int bb = i / w, j = i % w;
  out[i] = (j < da) ? a[bb * da + j] : b[bb * db + (j - da)];
}

__global__ void concat3_kernel(const float* __restrict__ a, const float* __restrict__ b,
                               const float* __restrict__ c, float* __restrict__ out) {
  int n = BATCH * 3 * DMODEL;
  int i = blockIdx.x * blockDim.x + threadIdx.x;
  if (i >= n) return;
  int bb = i / (3 * DMODEL), j = i % (3 * DMODEL);
  int sel = j / DMODEL, jj = j % DMODEL;
  const float* s = (sel == 0) ? a : (sel == 1) ? b : c;
  out[i] = s[bb * DMODEL + jj];
}

__device__ inline float sigmoidf_(float x) { return 1.0f / (1.0f + __expf(-x)); }

__global__ void gru_gates_kernel(const float* __restrict__ gi, const float* __restrict__ gh,
                                 float* __restrict__ h) {
  int i = blockIdx.x * blockDim.x + threadIdx.x;
  if (i >= BATCH * DMODEL) return;
  int b = i / DMODEL, j = i % DMODEL;
  const float* gib = gi + b * 3 * DMODEL;
  const float* ghb = gh + b * 3 * DMODEL;
  float r = sigmoidf_(gib[j] + ghb[j]);
  float z = sigmoidf_(gib[DMODEL + j] + ghb[DMODEL + j]);
  float n = tanhf(gib[2 * DMODEL + j] + r * ghb[2 * DMODEL + j]);
  h[i] = (1.0f - z) * n + z * h[i];
}

// Attention: score -> softmax -> context (context read from f16 enc mirror,
// halving the per-step L2 stream; fp32 accumulate).
__global__ __launch_bounds__(256) void attn1_kernel(
    const float* __restrict__ pre1, const float* __restrict__ q1,
    const float* __restrict__ vvec, const half_t* __restrict__ enc_h,
    float* __restrict__ ctx1, float* __restrict__ accum,
    float* __restrict__ copy_pred, int t) {
  const int b = blockIdx.x;
  const int tid = threadIdx.x;
  __shared__ float qs[DKDIM];
  __shared__ float vs[DKDIM];
  __shared__ float es[512];
  __shared__ float red[256];

  if (tid < DKDIM) { qs[tid] = q1[b * DKDIM + tid]; vs[tid] = vvec[tid]; }
  __syncthreads();

  for (int l = tid; l < LNODES; l += 256) {
    const float* p = pre1 + (long)(b * LNODES + l) * DKDIM;
    float s = 0.0f;
#pragma unroll 8
    for (int k = 0; k < DKDIM; ++k) s += vs[k] * tanhf(p[k] + qs[k]);
    es[l] = s;
  }
  __syncthreads();

  float m = -3.4e38f;
  for (int l = tid; l < LNODES; l += 256) m = fmaxf(m, es[l]);
  red[tid] = m; __syncthreads();
  for (int s = 128; s > 0; s >>= 1) {
    if (tid < s) red[tid] = fmaxf(red[tid], red[tid + s]);
    __syncthreads();
  }
  m = red[0]; __syncthreads();

  float sum = 0.0f;
  for (int l = tid; l < LNODES; l += 256) {
    float e = __expf(es[l] - m);
    es[l] = e; sum += e;
  }
  red[tid] = sum; __syncthreads();
  for (int s = 128; s > 0; s >>= 1) {
    if (tid < s) red[tid] += red[tid + s];
    __syncthreads();
  }
  const float inv = 1.0f / red[0];
  __syncthreads();

  for (int l = tid; l < LNODES; l += 256) {
    float a = es[l] * inv;
    es[l] = a;
    accum[b * LNODES + l] += a;
    copy_pred[(long)(b * TSTEPS + t) * LNODES + l] = a;
  }
  __syncthreads();

  for (int d = tid; d < DMODEL; d += 256) {
    const half_t* e = enc_h + (long)b * LNODES * DMODEL + d;
    float s = 0.0f;
    for (int l = 0; l < LNODES; ++l) s += es[l] * (float)e[(long)l * DMODEL];
    ctx1[b * DMODEL + d] = s;
  }
}

// g = sigmoid(gl); ctx = g*ctx1 + (1-g)*ctx2   (ctx2 == ans_node_src, exact identity)
__global__ void gatemix_kernel(const float* __restrict__ gl, const float* __restrict__ ctx1,
                               const float* __restrict__ ctx2, float* __restrict__ ctx) {
  int i = blockIdx.x * blockDim.x + threadIdx.x;
  if (i >= BATCH * DMODEL) return;
  float g = sigmoidf_(gl[i]);
  ctx[i] = g * ctx1[i] + (1.0f - g) * ctx2[i];
}

__global__ __launch_bounds__(256) void copygate_kernel(
    const float* __restrict__ h, const float* __restrict__ ctx,
    const float* __restrict__ csW, const float* __restrict__ csb,
    float* __restrict__ out, int t) {
  const int b = blockIdx.x;
  const int tid = threadIdx.x;
  __shared__ float red[256];
  float s = 0.0f;
  for (int j = tid; j < 2 * DMODEL; j += 256) {
    float x = (j < DMODEL) ? h[b * DMODEL + j] : ctx[b * DMODEL + (j - DMODEL)];
    s += x * csW[j];
  }
  red[tid] = s; __syncthreads();
  for (int k = 128; k > 0; k >>= 1) {
    if (tid < k) red[tid] += red[tid + k];
    __syncthreads();
  }
  if (tid == 0) out[b * TSTEPS + t] = sigmoidf_(red[0] + csb[0]);
}

// maxout pool-2 + transpose; emit f16 directly (consumed only by pred GEMM)
__global__ void maxout_kernel(const float* __restrict__ r, half_t* __restrict__ dec_out_h,
                              int t) {
  int i = blockIdx.x * blockDim.x + threadIdx.x;
  if (i >= BATCH * DHALF) return;
  int b = i / DHALF, j = i % DHALF;
  float a = r[b * DMODEL + 2 * j];
  float c = r[b * DMODEL + 2 * j + 1];
  dec_out_h[(long)(b * TSTEPS + t) * DHALF + j] = (half_t)fmaxf(a, c);
}

__global__ __launch_bounds__(256) void attnscore_kernel(const float* __restrict__ acc,
                                                        float* __restrict__ out) {
  const int b = blockIdx.x;
  const int tid = threadIdx.x;
  __shared__ float red[256];
  float s = 0.0f;
  for (int l = tid; l < LNODES; l += 256) s += acc[b * LNODES + l];
  red[tid] = s; __syncthreads();
  for (int k = 128; k > 0; k >>= 1) {
    if (tid < k) red[tid] += red[tid + k];
    __syncthreads();
  }
  const float inv = 1.0f / (red[0] + 1e-8f);
  __syncthreads();
  for (int l = tid; l < LNODES; l += 256) out[b * LNODES + l] = acc[b * LNODES + l] * inv;
}

__global__ void lasta_kernel(const float* __restrict__ copy_pred, float* __restrict__ out) {
  int i = blockIdx.x * blockDim.x + threadIdx.x;
  if (i >= BATCH * LNODES) return;
  int b = i / LNODES, l = i % LNODES;
  out[i] = copy_pred[(long)(b * TSTEPS + TSTEPS - 1) * LNODES + l];
}

// ---------------------------------------------------------------------------
extern "C" void kernel_launch(void* const* d_in, const int* in_sizes, int n_in,
                              void* d_out, int out_size, void* d_ws, size_t ws_size,
                              hipStream_t stream) {
  (void)in_sizes; (void)n_in; (void)out_size; (void)ws_size;
  const int*   tgt     = (const int*)d_in[0];
  const float* concat1 = (const float*)d_in[1];   // [B, L, 768]
  const float* ans     = (const float*)d_in[2];   // [B, D]
  const float* wemb    = (const float*)d_in[3];   // [V, D]
  const float* m_W     = (const float*)d_in[4];   // [D, 768]
  const float* m_b     = (const float*)d_in[5];
  const float* di_W    = (const float*)d_in[6];   // [D, D]
  const float* di_b    = (const float*)d_in[7];
  const float* Wih     = (const float*)d_in[8];   // [3D, 2D]
  const float* Whh     = (const float*)d_in[9];   // [3D, D]
  const float* bih     = (const float*)d_in[10];
  const float* bhh     = (const float*)d_in[11];
  const float* pre_W   = (const float*)d_in[12];  // [DK, D]
  const float* pre_b   = (const float*)d_in[13];
  const float* q_W     = (const float*)d_in[14];  // [DK, D]
  const float* vvec    = (const float*)d_in[15];  // [DK]
  const float* W2_W    = (const float*)d_in[16];  // [D, 3D]
  const float* W2_b    = (const float*)d_in[17];
  const float* cs_W    = (const float*)d_in[18];  // [1, 2D]
  const float* cs_b    = (const float*)d_in[19];
  const float* ro_W    = (const float*)d_in[20];  // [D, 3D]
  const float* ro_b    = (const float*)d_in[21];
  const float* W3_W    = (const float*)d_in[22];  // [D, D/2]
  const float* W3_b    = (const float*)d_in[23];

  // Output layout (flat, in reference return order)
  float* out = (float*)d_out;
  float* o_pred      = out;                                   // [B,T,D]
  float* o_a1last    = o_pred + (long)BATCH * TSTEPS * DMODEL;// [B,L]
  float* o_ctx       = o_a1last + (long)BATCH * LNODES;       // [B,D]
  float* o_attnsc    = o_ctx + (long)BATCH * DMODEL;          // [B,L]
  float* o_copy_pred = o_attnsc + (long)BATCH * LNODES;       // [B,T,L]
  float* o_copy_gate = o_copy_pred + (long)BATCH * TSTEPS * LNODES; // [B,T,1]

  // Workspace layout (bytes; f32 and f16 regions)
  char* wsb = (char*)d_ws;
  size_t off = 0;
  auto allocf = [&](size_t n) {
    float* p = (float*)(wsb + off);
    off = (off + n * sizeof(float) + 255) & ~(size_t)255;
    return p;
  };
  auto alloch = [&](size_t n) {
    half_t* p = (half_t*)(wsb + off);
    off = (off + n * sizeof(half_t) + 255) & ~(size_t)255;
    return p;
  };
  const int ML = BATCH * LNODES;                    // 32000
  float*  enc       = allocf((size_t)ML * DMODEL);  // [B*L, 512] fp32 master
  float*  pre1      = allocf((size_t)ML * DKDIM);   // [B*L, 64]
  float*  emb       = allocf((size_t)TSTEPS * BATCH * DMODEL);
  float*  accum     = allocf((size_t)ML);
  float*  h         = allocf((size_t)BATCH * DMODEL);
  float*  ctx       = allocf((size_t)BATCH * DMODEL);
  float*  ctx1      = allocf((size_t)BATCH * DMODEL);
  float*  xbuf      = allocf((size_t)BATCH * 2 * DMODEL);
  float*  cat3      = allocf((size_t)BATCH * 3 * DMODEL);
  float*  gi        = allocf((size_t)BATCH * 3 * DMODEL);
  float*  gh        = allocf((size_t)BATCH * 3 * DMODEL);
  float*  q1        = allocf((size_t)BATCH * DKDIM);
  float*  gbuf      = allocf((size_t)BATCH * DMODEL);
  float*  rbuf      = allocf((size_t)BATCH * DMODEL);
  half_t* concat1_h = alloch((size_t)ML * ENCIN);   // f16 feed for async enc GEMM
  half_t* m_W_h     = alloch((size_t)DMODEL * ENCIN);
  half_t* enc_h     = alloch((size_t)ML * DMODEL);  // f16 mirror for attention
  half_t* dec_out_h = alloch((size_t)BATCH * TSTEPS * DHALF);
  half_t* W3_W_h    = alloch((size_t)DMODEL * DHALF);

  const int TPB = 256;
  auto blocks = [](long n) { return (unsigned)((n + 255) / 256); };

  // --- Precompute ---------------------------------------------------------
  embed_kernel<<<blocks((long)TSTEPS * BATCH * DMODEL), TPB, 0, stream>>>(
      tgt, wemb, emb, TSTEPS * BATCH * DMODEL);

  cast_h_kernel<<<blocks((long)ML * ENCIN), TPB, 0, stream>>>(
      concat1, concat1_h, (long)ML * ENCIN);
  cast_h_kernel<<<blocks((long)DMODEL * ENCIN), TPB, 0, stream>>>(
      m_W, m_W_h, (long)DMODEL * ENCIN);
  cast_h_kernel<<<blocks((long)DMODEL * DHALF), TPB, 0, stream>>>(
      W3_W, W3_W_h, (long)DMODEL * DHALF);

  // enc = concatnodes1 @ m_W^T + m_b   [32000 x 512], K=768  (async pipeline)
  wmma_gemm_h_kernel<<<dim3(DMODEL / 64, ML / 64), TPB, 0, stream>>>(
      concat1_h, m_W_h, m_b, enc, enc_h, ML, DMODEL, ENCIN);

  // pre1 = enc @ pre_W^T + pre_b       [32000 x 64], K=512
  wmma_gemm_kernel<0><<<dim3(DKDIM / 64, ML / 64), TPB, 0, stream>>>(
      enc, pre_W, pre_b, pre1, ML, DKDIM, DMODEL);

  // h0 = tanh(ans @ di_W^T + di_b)     [64 x 512]
  wmma_gemm_kernel<1><<<dim3(DMODEL / 64, BATCH / 64), TPB, 0, stream>>>(
      ans, di_W, di_b, h, BATCH, DMODEL, DMODEL);

  zero_kernel<<<blocks(BATCH * DMODEL), TPB, 0, stream>>>(ctx, BATCH * DMODEL);
  zero_kernel<<<blocks(ML), TPB, 0, stream>>>(accum, ML);

  // --- Decoder timestep loop ---------------------------------------------
  for (int t = 0; t < TSTEPS; ++t) {
    const float* e_t = emb + (size_t)t * BATCH * DMODEL;

    concat2_kernel<<<blocks(BATCH * 2 * DMODEL), TPB, 0, stream>>>(
        e_t, ctx, xbuf, DMODEL, DMODEL);

    wmma_gemm_kernel<0><<<dim3(3 * DMODEL / 64, 1), TPB, 0, stream>>>(
        xbuf, Wih, bih, gi, BATCH, 3 * DMODEL, 2 * DMODEL);
    wmma_gemm_kernel<0><<<dim3(3 * DMODEL / 64, 1), TPB, 0, stream>>>(
        h, Whh, bhh, gh, BATCH, 3 * DMODEL, DMODEL);

    gru_gates_kernel<<<blocks(BATCH * DMODEL), TPB, 0, stream>>>(gi, gh, h);

    wmma_gemm_kernel<0><<<dim3(1, 1), TPB, 0, stream>>>(
        h, q_W, nullptr, q1, BATCH, DKDIM, DMODEL);

    attn1_kernel<<<BATCH, TPB, 0, stream>>>(pre1, q1, vvec, enc_h, ctx1, accum,
                                            o_copy_pred, t);

    // ctx2 == ans (identity: uniform softmax over broadcast context)
    concat3_kernel<<<blocks(BATCH * 3 * DMODEL), TPB, 0, stream>>>(h, ctx1, ans, cat3);
    wmma_gemm_kernel<0><<<dim3(DMODEL / 64, 1), TPB, 0, stream>>>(
        cat3, W2_W, W2_b, gbuf, BATCH, DMODEL, 3 * DMODEL);
    gatemix_kernel<<<blocks(BATCH * DMODEL), TPB, 0, stream>>>(gbuf, ctx1, ans, ctx);

    copygate_kernel<<<BATCH, TPB, 0, stream>>>(h, ctx, cs_W, cs_b, o_copy_gate, t);

    concat3_kernel<<<blocks(BATCH * 3 * DMODEL), TPB, 0, stream>>>(e_t, h, ctx, cat3);
    wmma_gemm_kernel<0><<<dim3(DMODEL / 64, 1), TPB, 0, stream>>>(
        cat3, ro_W, ro_b, rbuf, BATCH, DMODEL, 3 * DMODEL);

    maxout_kernel<<<blocks(BATCH * DHALF), TPB, 0, stream>>>(rbuf, dec_out_h, t);
  }

  // --- Final projections / outputs ---------------------------------------
  // pred = dec_out @ W3^T + W3_b   [4096 x 512], K=256  (async pipeline)
  wmma_gemm_h_kernel<<<dim3(DMODEL / 64, BATCH * TSTEPS / 64), TPB, 0, stream>>>(
      dec_out_h, W3_W_h, W3_b, o_pred, nullptr, BATCH * TSTEPS, DMODEL, DHALF);

  attnscore_kernel<<<BATCH, TPB, 0, stream>>>(accum, o_attnsc);
  lasta_kernel<<<blocks(ML), TPB, 0, stream>>>(o_copy_pred, o_a1last);
  copy_kernel<<<blocks(BATCH * DMODEL), TPB, 0, stream>>>(ctx, o_ctx, BATCH * DMODEL);
}
